// Varkeys_55602646614102
// MI455X (gfx1250) — matrix operand order repaired and checked by reading, not compiled, not run on hardware
//
#include <hip/hip_runtime.h>

// ---------------------------------------------------------------------------
// Varkeys fused kernel for MI455X (gfx1250, wave32, WMMA bf16 16x16x32)
//   S' = (keys_bf16) @ (-2*x_bf16)^T          (GEMM1, per 32x16 tile)
//   kern = 1 / (S' + kk + xx + EPS)           (register-resident)
//   out += kern^T @ values                    (GEMM2, fused, K = dict dim)
// Keys tiles staged into double-buffered LDS via the Tensor Data Mover
// (tensor_load_to_lds), overlapped with WMMA compute of the previous tile.
// ---------------------------------------------------------------------------

typedef __attribute__((ext_vector_type(16))) __bf16 v16bf;
typedef __attribute__((ext_vector_type(8)))  float  v8f;
typedef __attribute__((ext_vector_type(4)))  unsigned int u32x4;
typedef __attribute__((ext_vector_type(4)))  int i32x4;
typedef __attribute__((ext_vector_type(8)))  int i32x8;

#define EPS 1e-4f

#define BATCH 4096
#define KEYSZ 512
#define DICT  16384
#define CATS  100
#define CPAD  112          // 7 tiles of 16 categories (padded with zeros)

#define NDSPLIT 8
#define DSLICE  (DICT / NDSPLIT)   // 2048
#define WG_BATCH 128               // 8 waves x 16 batch rows

// LDS layout: padded rows (520 bf16 = 1040 B) to spread rows across banks.
// TDM reproduces the padding: pad_interval=7 (256 DW = 1024 B), pad_amount=3 (4 DW = 16 B).
#define ROW_B   1040
#define XSLAB_B (WG_BATCH * ROW_B)         // 133120
#define KTILE_OFF XSLAB_B
#define KTILE_B (32 * ROW_B)               // 33280 per buffer
#define SMEM_BYTES (XSLAB_B + 2 * KTILE_B) // 199680 (< 320 KB WGP LDS)

// workspace layout (bytes)
#define WS_KB   0u                          // keys bf16:   16384*512*2 = 16 MB
#define WS_XB   16777216u                   // x bf16*-2:    4096*512*2 =  4 MB
#define WS_VT   20971520u                   // valuesT bf16:  112*16384*2
#define WS_KK   24641536u                   // kk f32: 16384
#define WS_XX   24707072u                   // xx f32: 4096
#define WS_PART 24723456u                   // partials: 8*4096*112 f32

#if defined(__gfx1250__) && __has_builtin(__builtin_amdgcn_tensor_load_to_lds) && \
    __has_builtin(__builtin_amdgcn_s_wait_tensorcnt)
#define USE_TDM 1
#else
#define USE_TDM 0
#endif

__device__ __forceinline__ unsigned f2bf_bits(float f) {
  unsigned u = __float_as_uint(f);
  return (u + 0x7FFFu + ((u >> 16) & 1u)) >> 16;   // RNE f32 -> bf16
}

struct U128x2 { uint4 lo, hi; };
__device__ __forceinline__ v16bf make_v16bf(uint4 lo, uint4 hi) {
  U128x2 t{lo, hi};
  return __builtin_bit_cast(v16bf, t);
}

#if USE_TDM
// Issue one TDM load: 2D tile (32 rows x 512 bf16) of kb starting at row d0
// into LDS byte offset lds_off, with 16 B of LDS padding appended per 1024 B row.
__device__ __forceinline__ void tdm_load_ktile(const unsigned short* kb, int d0,
                                               unsigned lds_off) {
  unsigned long long ga = (unsigned long long)(size_t)(kb + (size_t)d0 * KEYSZ);
  // D# group 0: count=1 | lds_addr | global_addr[56:0] | type=2 (bits 127:126)
  u32x4 g0 = { 1u,
               lds_off,
               (unsigned)(ga & 0xFFFFFFFFull),
               (unsigned)((ga >> 32) & 0x1FFFFFFull) | (2u << 30) };
  // D# group 1:
  //  dw0: wg_mask=0, data_size=1(2B), pad_enable(bit20), pad_interval=7(b24:22),
  //       pad_amount=3(b31:25)
  //  dw1: [31:16]=tensor_dim0.lo=512     dw2: [31:16]=tensor_dim1.lo=16384
  //  dw3: [31:16]=tile_dim0=512          dw4: [15:0]=tile_dim1=32
  //  dw5: tensor_dim0_stride.lo=512      dw6,dw7: 0
  i32x8 g1 = { (int)((1u << 16) | (1u << 20) | (7u << 22) | (3u << 25)),
               (int)(512u << 16),
               (int)(16384u << 16),
               (int)(512u << 16),
               32,
               512,
               0, 0 };
  i32x4 z4 = {0, 0, 0, 0};
#if __clang_major__ >= 23
  i32x8 z8 = {0, 0, 0, 0, 0, 0, 0, 0};
  __builtin_amdgcn_tensor_load_to_lds(g0, g1, z4, z4, z8, 0);
#else
  __builtin_amdgcn_tensor_load_to_lds(g0, g1, z4, z4, 0);
#endif
}
#endif

// --- prep: per-row bf16 convert (with scale) + f32 sum of squares ----------
__global__ void rowprep_kernel(const float* __restrict__ src,
                               unsigned short* __restrict__ dstB,
                               float* __restrict__ dstS,
                               int nrows, float scale) {
  const int wv = threadIdx.x >> 5, lane = threadIdx.x & 31;
  const int row = blockIdx.x * 8 + wv;
  if (row >= nrows) return;
  const float* s = src + (size_t)row * KEYSZ;
  unsigned short* d = dstB + (size_t)row * KEYSZ;
  float acc = 0.f;
#pragma unroll
  for (int j = 0; j < KEYSZ / 32; ++j) {
    const int col = j * 32 + lane;
    const float v = s[col];
    acc += v * v;
    d[col] = (unsigned short)f2bf_bits(v * scale);
  }
#pragma unroll
  for (int off = 16; off > 0; off >>= 1) acc += __shfl_down(acc, off, 32);
  if (lane == 0) dstS[row] = acc;
}

// --- prep: values (D,100) -> bf16 valuesT (112,D), zero-padded -------------
__global__ void valtrans_kernel(const float* __restrict__ values,
                                unsigned short* __restrict__ vT) {
  const int gid = blockIdx.x * 256 + threadIdx.x;   // CPAD*DICT threads
  const int d = gid & (DICT - 1);
  const int c = gid >> 14;
  const float v = (c < CATS) ? values[(size_t)d * CATS + c] : 0.f;
  vT[(size_t)c * DICT + d] = (unsigned short)f2bf_bits(v);
}

// --- main fused kernel -----------------------------------------------------
__global__ __launch_bounds__(256) void varkeys_main(
    const unsigned short* __restrict__ kb,
    const unsigned short* __restrict__ xb,
    const unsigned short* __restrict__ vT,
    const float* __restrict__ kk,
    const float* __restrict__ xx,
    float* __restrict__ part) {
  extern __shared__ uint4 smem4[];
  char* smem = (char*)smem4;

  const int tid  = threadIdx.x;
  const int wv   = tid >> 5, lane = tid & 31;
  const int half = lane >> 4, l16 = lane & 15;
  const int bbase = blockIdx.x * WG_BATCH;
  const int dbase = blockIdx.y * DSLICE;
  const int nTiles = DSLICE / 32;

  // Stage this WG's x slab (128 rows x 512 bf16) into padded LDS rows.
  for (int idx = tid; idx < WG_BATCH * 64; idx += 256) {
    const int row = idx >> 6, ch = idx & 63;
    uint4 v = *(const uint4*)(xb + (size_t)(bbase + row) * KEYSZ + ch * 8);
    *(uint4*)(smem + row * ROW_B + ch * 16) = v;
  }

#if USE_TDM
  if (wv == 0) tdm_load_ktile(kb, dbase, KTILE_OFF);   // prime buffer 0
#endif

  const float xxl = xx[bbase + wv * 16 + l16];

  v8f acc[7];
#pragma unroll
  for (int i = 0; i < 7; ++i) acc[i] = (v8f)0.0f;

  const char* xrow_base = smem + (wv * 16 + l16) * ROW_B;

  for (int dt = 0; dt < nTiles; ++dt) {
    const int d0 = dbase + dt * 32;

#if USE_TDM
    if (wv == 0) __builtin_amdgcn_s_wait_tensorcnt(0);  // tile dt landed
    __syncthreads();  // tile dt visible to all; prev reads of other buf done
    if (wv == 0 && dt + 1 < nTiles)                     // overlap next DMA
      tdm_load_ktile(kb, d0 + 32, KTILE_OFF + (unsigned)((dt + 1) & 1) * KTILE_B);
    const char* ktile = smem + KTILE_OFF + (size_t)(dt & 1) * KTILE_B;
#else
    __syncthreads();  // previous tile fully consumed
    for (int idx = tid; idx < 32 * 64; idx += 256) {
      const int row = idx >> 6, ch = idx & 63;
      uint4 v = *(const uint4*)(kb + (size_t)(d0 + row) * KEYSZ + ch * 8);
      *(uint4*)(smem + KTILE_OFF + row * ROW_B + ch * 16) = v;
    }
    __syncthreads();  // keys tile ready
    const char* ktile = smem + KTILE_OFF;
#endif
    const char* karow = ktile + l16 * ROW_B;
    const char* kbrow = karow + 16 * ROW_B;

    // Pull next tile's valuesT rows toward L1 while GEMM1 runs.
    if (dt + 1 < nTiles) {
      const unsigned short* vn = vT + (size_t)l16 * DICT + (d0 + 32) + 16 * half;
#pragma unroll
      for (int ct = 0; ct < 7; ++ct)
        __builtin_prefetch(vn + (size_t)(ct * 16) * DICT, 0, 0);
    }

    // GEMM1: S' (32 dict x 16 batch) = keys_tile @ (-2 x_tile)^T
    v8f S0 = (v8f)0.0f, S1 = (v8f)0.0f;
#pragma unroll
    for (int ks = 0; ks < KEYSZ / 32; ++ks) {
      const int k0 = ks * 32;
      // B (32x16): lane n=l16 holds K = 8*half+0..7 and 16+8*half+0..7
      const char* xp = xrow_base + (k0 + 8 * half) * 2;
      v16bf B1 = make_v16bf(*(const uint4*)xp, *(const uint4*)(xp + 32));
      // A (16x32): lane m=l16 holds K = 8*half+0..7 and 16+8*half+0..7
      const char* ap = karow + (k0 + 8 * half) * 2;
      v16bf A1a = make_v16bf(*(const uint4*)ap, *(const uint4*)(ap + 32));
      const char* bp = kbrow + (k0 + 8 * half) * 2;
      v16bf A1b = make_v16bf(*(const uint4*)bp, *(const uint4*)(bp + 32));
      S0 = __builtin_amdgcn_wmma_f32_16x16x32_bf16(false, A1a, false, B1,
                                                   (short)0, S0, false, false);
      S1 = __builtin_amdgcn_wmma_f32_16x16x32_bf16(false, A1b, false, B1,
                                                   (short)0, S1, false, false);
    }

    // kern = 1/(S' + kk + xx + EPS)
    const float* kk0p = kk + d0 + 8 * half;
    const float4 kA0 = *(const float4*)(kk0p);
    const float4 kA1 = *(const float4*)(kk0p + 4);
    const float4 kB0 = *(const float4*)(kk0p + 16);
    const float4 kB1 = *(const float4*)(kk0p + 20);
    const float kka[8] = {kA0.x, kA0.y, kA0.z, kA0.w, kA1.x, kA1.y, kA1.z, kA1.w};
    const float kkb[8] = {kB0.x, kB0.y, kB0.z, kB0.w, kB1.x, kB1.y, kB1.z, kB1.w};
    float kern0[8], kern1[8];
#pragma unroll
    for (int r = 0; r < 8; ++r) {
      kern0[r] = 1.0f / (S0[r] + kka[r] + xxl + EPS);
      kern1[r] = 1.0f / (S1[r] + kkb[r] + xxl + EPS);
    }

    // C-layout (lane=batch, vgpr=dict) -> A-layout (16 batch x 32 dict):
    // pure per-lane pack, no cross-lane movement needed.
    unsigned a2[8];
#pragma unroll
    for (int j = 0; j < 4; ++j) {
      a2[j]     = f2bf_bits(kern0[2 * j]) | (f2bf_bits(kern0[2 * j + 1]) << 16);
      a2[4 + j] = f2bf_bits(kern1[2 * j]) | (f2bf_bits(kern1[2 * j + 1]) << 16);
    }
    const v16bf A2 = make_v16bf(make_uint4(a2[0], a2[1], a2[2], a2[3]),
                                make_uint4(a2[4], a2[5], a2[6], a2[7]));

    // GEMM2: out_tile(16 batch x 112 cat) += kern^T @ valuesT tiles
    const unsigned short* vbase = vT + (size_t)l16 * DICT + d0 + 8 * half;
#pragma unroll
    for (int ct = 0; ct < 7; ++ct) {
      const unsigned short* vp = vbase + (size_t)(ct * 16) * DICT;
      v16bf B2 = make_v16bf(*(const uint4*)vp, *(const uint4*)(vp + 16));
      acc[ct] = __builtin_amdgcn_wmma_f32_16x16x32_bf16(false, A2, false, B2,
                                                        (short)0, acc[ct], false, false);
    }
  }

  // write f32 partials (deterministic; reduced by reduce_kernel)
  const int brow0 = bbase + wv * 16 + 8 * half;
  float* pout = part + (size_t)blockIdx.y * BATCH * CPAD;
#pragma unroll
  for (int ct = 0; ct < 7; ++ct) {
    const int c = ct * 16 + l16;
#pragma unroll
    for (int r = 0; r < 8; ++r)
      pout[(size_t)(brow0 + r) * CPAD + c] = acc[ct][r];
  }
}

// --- deterministic reduction over the 8 dict-splits ------------------------
__global__ void reduce_kernel(const float* __restrict__ part,
                              float* __restrict__ out) {
  const int gid = blockIdx.x * 256 + threadIdx.x;
  if (gid >= BATCH * CATS) return;
  const int b = gid / CATS, c = gid % CATS;
  float s = 0.f;
#pragma unroll
  for (int ds = 0; ds < NDSPLIT; ++ds)
    s += part[((size_t)ds * BATCH + b) * CPAD + c];
  out[gid] = s;
}

extern "C" void kernel_launch(void* const* d_in, const int* in_sizes, int n_in,
                              void* d_out, int out_size, void* d_ws, size_t ws_size,
                              hipStream_t stream) {
  const float* x      = (const float*)d_in[0];   // (4096, 512)
  const float* keys   = (const float*)d_in[1];   // (16384, 512)
  const float* values = (const float*)d_in[2];   // (16384, 100)
  float* out = (float*)d_out;

  char* ws = (char*)d_ws;
  unsigned short* kb = (unsigned short*)(ws + WS_KB);
  unsigned short* xb = (unsigned short*)(ws + WS_XB);
  unsigned short* vT = (unsigned short*)(ws + WS_VT);
  float* kk   = (float*)(ws + WS_KK);
  float* xx   = (float*)(ws + WS_XX);
  float* part = (float*)(ws + WS_PART);

  rowprep_kernel<<<DICT / 8, 256, 0, stream>>>(keys, kb, kk, DICT, 1.0f);
  rowprep_kernel<<<BATCH / 8, 256, 0, stream>>>(x, xb, xx, BATCH, -2.0f);
  valtrans_kernel<<<(CPAD * DICT) / 256, 256, 0, stream>>>(values, vT);

  dim3 grid(BATCH / WG_BATCH, NDSPLIT);  // (32, 8)
  varkeys_main<<<grid, 256, SMEM_BYTES, stream>>>(kb, xb, vT, kk, xx, part);

  reduce_kernel<<<(BATCH * CATS + 255) / 256, 256, 0, stream>>>(part, out);
}
